// RNNLanguageModel_3272765080017
// MI455X (gfx1250) — matrix operand order repaired and checked by reading, not compiled, not run on hardware
//
#include <hip/hip_runtime.h>
#include <hip/hip_bf16.h>

// ---------------------------------------------------------------------------
// RNN LM (2-layer GRU, H=1536, B=256, L=128) for MI455X / gfx1250.
// bf16 WMMA (V_WMMA_F32_16X16X32_BF16), f32 accumulate, f32 master h state.
// Weight (B) tiles are staged per-workgroup into LDS with CDNA5 async
// global->LDS copies (ASYNCcnt, double-buffered, one barrier per 64-K stage)
// and reused by all waves; each wave computes 32x16 of h_new (2 M-tiles) so B
// fragments amortize over 6 WMMAs. All per-lane global accesses are forced to
// address-space 1 so they lower to global_load/store (not flat_*, which would
// contend with the DS pipe used by the LDS fragment reads).
// ~51 MB of d_ws required.
// ---------------------------------------------------------------------------

#define VOCAB 128
#define EMB   128
#define HID   1536
#define BATCH 256
#define SEQ   128
#define H3    (3 * HID)
#define BSTR  72   // LDS row stride in bf16 elems (64 data + 8 pad)

typedef __attribute__((ext_vector_type(16))) __bf16 v16bf;
typedef __attribute__((ext_vector_type(8)))  float  v8f;
// must match the async builtin's parameter type exactly (gcc-vector int4)
typedef __attribute__((__vector_size__(4 * sizeof(int)))) int v4i;
// trivially-copyable 16B vector (HIP's uint4 is a class and can't be copied
// from an address-space-1 lvalue)
typedef __attribute__((__vector_size__(16))) unsigned int u32x4;

union Frag {
    u32x4 q[2];     // two 16-byte chunks (8 bf16 each)
    v16bf v;        // 16 bf16 = 8 VGPRs, WMMA operand
};

// ---- force global (AS1) lowering for per-lane global accesses -------------
template <typename T>
__device__ __forceinline__ T loadg(const T* p) {
    return *(const __attribute__((address_space(1))) T*)p;
}
template <typename T>
__device__ __forceinline__ void storeg(T* p, T v) {
    *(__attribute__((address_space(1))) T*)p = v;
}

// ---- CDNA5 async global->LDS copy (ASYNCcnt) with portable fallback -------
#if __has_builtin(__builtin_amdgcn_global_load_async_to_lds_b128)
#define ASYNC_CP16(SRC, DST)                                                \
    __builtin_amdgcn_global_load_async_to_lds_b128(                         \
        (__attribute__((address_space(1))) v4i*)(SRC),                      \
        (__attribute__((address_space(3))) v4i*)(DST), 0, 0)
#if __has_builtin(__builtin_amdgcn_s_wait_asynccnt)
#define WAIT_ASYNC() __builtin_amdgcn_s_wait_asynccnt(0)
#else
#define WAIT_ASYNC() asm volatile("s_wait_asynccnt 0x0" ::: "memory")
#endif
#else
// Fallback: synchronous copy through VGPRs (ds_store); barrier handles waits.
#define ASYNC_CP16(SRC, DST) (*(u32x4*)(DST) = *(const u32x4*)(SRC))
#define WAIT_ASYNC() ((void)0)
#endif

__device__ __forceinline__ unsigned short f2bf(float f) {
    unsigned int u = __float_as_uint(f);
    u += 0x7FFFu + ((u >> 16) & 1u);      // round-to-nearest-even
    return (unsigned short)(u >> 16);
}

__device__ __forceinline__ float sigmoid_(float x) {
    return 1.0f / (1.0f + __expf(-x));
}

__device__ __forceinline__ v8f wmma_bf16(const Frag& a, const Frag& b, v8f c) {
    return __builtin_amdgcn_wmma_f32_16x16x32_bf16(false, a.v, false, b.v,
                                                   (short)0, c, false, false);
}

// Load a 16x32 bf16 A-fragment: lane holds row `rowptr`, K-chunks at
// kb+half*8 and kb+16+half*8 (ISA 16-bit A layout), via global_load_b128.
__device__ __forceinline__ Frag load_a_frag(const unsigned short* rowptr,
                                            int kb, int half) {
    Frag f;
    f.q[0] = loadg((const u32x4*)(rowptr + kb + half * 8));
    f.q[1] = loadg((const u32x4*)(rowptr + kb + 16 + half * 8));
    return f;
}

// Load a 32x16 bf16 B-fragment from an LDS row (row = output col, K-chunks
// at p and p+16, p = kb+half*8) -> two ds_load_b128.
__device__ __forceinline__ Frag load_b_frag_lds(const unsigned short* p) {
    Frag f;
    f.q[0] = *(const u32x4*)p;
    f.q[1] = *(const u32x4*)(p + 16);
    return f;
}

// Stage 6 weight tiles (3 gates x 2 N-subtiles), 16 rows x 64 K bf16 each,
// for K-span [ks0, ks0+64) into LDS buffer laid out [6][16][BSTR].
__device__ __forceinline__ void stage_w_tiles(
    const unsigned short* __restrict__ w, int Kdim, int n0blk, int ks0,
    unsigned short* dstbuf, int tid)
{
#pragma unroll
    for (int j = 0; j < 3; ++j) {
        const int c = tid + j * 256;           // 0..767 chunk id (16B chunks)
        const int g = c >> 7;                  // 0..5: gate*2 + nsub
        const int r = (c >> 3) & 15;           // weight row within tile
        const int p = c & 7;                   // 16B chunk within 128B row
        const unsigned short* src =
            w + (size_t)((g >> 1) * HID + n0blk + (g & 1) * 16 + r) * Kdim
              + ks0 + p * 8;
        unsigned short* dst = dstbuf + (g * 16 + r) * BSTR + p * 8;
        ASYNC_CP16(src, dst);
    }
}

// Stage 16 rows x 64 K bf16 of an activation matrix into LDS [16][BSTR].
__device__ __forceinline__ void stage_rows16(
    const unsigned short* __restrict__ src_base, int ks0,
    unsigned short* dstbuf, int tid)
{
    if (tid < 128) {
        const int r = tid >> 3, p = tid & 7;
        ASYNC_CP16(src_base + (size_t)r * HID + ks0 + p * 8,
                   dstbuf + r * BSTR + p * 8);
    }
}

// --------------------------- setup kernels ---------------------------------

__global__ void f32_to_bf16_kernel(const float* __restrict__ in,
                                   unsigned short* __restrict__ out, int n) {
    int i = blockIdx.x * blockDim.x + threadIdx.x;
    if (i < n) storeg(&out[i], f2bf(loadg(&in[i])));
}

__global__ void zero_h_kernel(float* __restrict__ h0f, float* __restrict__ h1f,
                              unsigned short* __restrict__ h0b,
                              unsigned short* __restrict__ h1b, int n) {
    int i = blockIdx.x * blockDim.x + threadIdx.x;
    if (i < n) {
        storeg(&h0f[i], 0.0f); storeg(&h1f[i], 0.0f);
        storeg(&h0b[i], (unsigned short)0); storeg(&h1b[i], (unsigned short)0);
    }
}

// --------------------------- GRU layer kernel ------------------------------
// Grid: (HID/32, BATCH/128), block 256 threads (8 waves).
// Wave w: rows m0 = by*128 + (w&3)*32 (two 16-row M-tiles),
//         cols n0 = bx*32 + (w>>2)*16.
// acc_r/acc_z take both GEMMs (i_* + h_*); i_n and h_n kept separate.

__global__ void __launch_bounds__(256) gru_layer_kernel(
    const unsigned short* __restrict__ xA,   // batch!=null: emb_bf; else A[BATCH][Kin]
    const int* __restrict__ batch, int t, int Kin,
    const unsigned short* __restrict__ w_ih, // bf16 [3H][Kin] row-major
    const unsigned short* __restrict__ w_hh, // bf16 [3H][HID] row-major
    const unsigned short* __restrict__ h_bf, // bf16 h_prev [BATCH][HID]
    const float*          __restrict__ h_f,  // f32  h_prev
    const float* __restrict__ b_ih,
    const float* __restrict__ b_hh,
    float*          __restrict__ hn_f,
    unsigned short* __restrict__ hn_bf)
{
    __shared__ __align__(16) unsigned short sB[2][6 * 16 * BSTR];

    const int tid  = threadIdx.x;
    const int lane = tid & 31;
    const int wave = tid >> 5;
    const int colL = lane & 15;
    const int half = lane >> 4;

    const int n0blk = blockIdx.x * 32;
    const int nsub  = wave >> 2;
    const int n0    = n0blk + nsub * 16;
    const int m0    = blockIdx.y * 128 + (wave & 3) * 32;

    v8f acc_r[2]  = {{}, {}};
    v8f acc_z[2]  = {{}, {}};
    v8f acc_in[2] = {{}, {}};
    v8f acc_hn[2] = {{}, {}};

    // A-row pointers (per lane: row m0 + mt*16 + colL)
    const unsigned short* arow[2];
    if (batch) {
        const int tok0 = loadg(&batch[(m0 + colL) * SEQ + t]);
        const int tok1 = loadg(&batch[(m0 + 16 + colL) * SEQ + t]);
        arow[0] = xA + (size_t)tok0 * EMB;
        arow[1] = xA + (size_t)tok1 * EMB;
    } else {
        arow[0] = xA + (size_t)(m0 + colL) * Kin;
        arow[1] = xA + (size_t)(m0 + 16 + colL) * Kin;
    }
    const unsigned short* hrow[2] = {
        h_bf + (size_t)(m0 + colL) * HID,
        h_bf + (size_t)(m0 + 16 + colL) * HID
    };

    // ================= phase 0: gi = x @ w_ih^T =================
    {
        const int S = Kin / 64;
        stage_w_tiles(w_ih, Kin, n0blk, 0, &sB[0][0], tid);
        for (int s = 0; s < S; ++s) {
            WAIT_ASYNC();
            __syncthreads();
            if (s + 1 < S)
                stage_w_tiles(w_ih, Kin, n0blk, (s + 1) * 64, &sB[(s + 1) & 1][0], tid);
            const unsigned short* bp = &sB[s & 1][0];
            const int ks0 = s * 64;
#pragma unroll
            for (int kk = 0; kk < 64; kk += 32) {
                const int boff = kk + half * 8;
                Frag fr = load_b_frag_lds(bp + ((0 + nsub) * 16 + colL) * BSTR + boff);
                Frag fz = load_b_frag_lds(bp + ((2 + nsub) * 16 + colL) * BSTR + boff);
                Frag fn = load_b_frag_lds(bp + ((4 + nsub) * 16 + colL) * BSTR + boff);
#pragma unroll
                for (int mt = 0; mt < 2; ++mt) {
                    Frag fa = load_a_frag(arow[mt], ks0 + kk, half);
                    acc_r[mt]  = wmma_bf16(fa, fr, acc_r[mt]);
                    acc_z[mt]  = wmma_bf16(fa, fz, acc_z[mt]);
                    acc_in[mt] = wmma_bf16(fa, fn, acc_in[mt]);
                }
            }
        }
    }

    // ================= phase 1: gh = h_prev @ w_hh^T =================
    {
        const int S = HID / 64;   // 24 (even, so buffer parity is safe)
        stage_w_tiles(w_hh, HID, n0blk, 0, &sB[0][0], tid);
        for (int s = 0; s < S; ++s) {
            WAIT_ASYNC();
            __syncthreads();
            if (s + 1 < S)
                stage_w_tiles(w_hh, HID, n0blk, (s + 1) * 64, &sB[(s + 1) & 1][0], tid);
            const unsigned short* bp = &sB[s & 1][0];
            const int ks0 = s * 64;
#pragma unroll
            for (int kk = 0; kk < 64; kk += 32) {
                const int boff = kk + half * 8;
                Frag fr = load_b_frag_lds(bp + ((0 + nsub) * 16 + colL) * BSTR + boff);
                Frag fz = load_b_frag_lds(bp + ((2 + nsub) * 16 + colL) * BSTR + boff);
                Frag fn = load_b_frag_lds(bp + ((4 + nsub) * 16 + colL) * BSTR + boff);
#pragma unroll
                for (int mt = 0; mt < 2; ++mt) {
                    Frag fa = load_a_frag(hrow[mt], ks0 + kk, half);
                    acc_r[mt]  = wmma_bf16(fa, fr, acc_r[mt]);
                    acc_z[mt]  = wmma_bf16(fa, fz, acc_z[mt]);
                    acc_hn[mt] = wmma_bf16(fa, fn, acc_hn[mt]);
                }
            }
        }
    }

    // ================= fused GRU gate epilogue =================
    // C/D layout: element v -> row m0 + mt*16 + v + 8*half, col n0 + colL.
    const int j = n0 + colL;
    const float bir = loadg(&b_ih[j]);
    const float biz = loadg(&b_ih[HID + j]);
    const float bin_ = loadg(&b_ih[2 * HID + j]);
    const float bhr = loadg(&b_hh[j]);
    const float bhz = loadg(&b_hh[HID + j]);
    const float bhn = loadg(&b_hh[2 * HID + j]);
#pragma unroll
    for (int mt = 0; mt < 2; ++mt) {
#pragma unroll
        for (int v = 0; v < 8; ++v) {
            const int row = m0 + mt * 16 + v + 8 * half;
            const float r = sigmoid_(acc_r[mt][v] + bir + bhr);
            const float z = sigmoid_(acc_z[mt][v] + biz + bhz);
            const float n = tanhf(acc_in[mt][v] + bin_ + r * (acc_hn[mt][v] + bhn));
            const float hp = loadg(&h_f[(size_t)row * HID + j]);
            const float hv = (1.0f - z) * n + z * hp;
            storeg(&hn_f[(size_t)row * HID + j], hv);
            storeg(&hn_bf[(size_t)row * HID + j], f2bf(hv));
        }
    }
}

// --------------------------- logits + log-softmax --------------------------
// Grid: BATCH/16 blocks, 256 threads. All 8 waves share the same 16 h1 rows,
// so A is async-staged in LDS; each wave owns 16 vocab cols (B from global).

__global__ void __launch_bounds__(256) logits_kernel(
    const unsigned short* __restrict__ h1_bf,  // [BATCH][HID]
    const unsigned short* __restrict__ outw,   // bf16 [VOCAB][HID]
    const float* __restrict__ out_b,
    const int*   __restrict__ batch, int t,
    float* __restrict__ lp_out)                // [BATCH][SEQ-1]
{
    __shared__ __align__(16) unsigned short sA[2][16 * BSTR];
    __shared__ float sm[16][VOCAB + 4];

    const int tid  = threadIdx.x;
    const int lane = tid & 31;
    const int wave = tid >> 5;
    const int m0   = blockIdx.x * 16;
    const int n0   = wave * 16;
    const int colL = lane & 15;
    const int half = lane >> 4;

    v8f acc = {};
    const unsigned short* abase = h1_bf + (size_t)m0 * HID;
    const unsigned short* brow  = outw + (size_t)(n0 + colL) * HID;

    const int S = HID / 64;
    stage_rows16(abase, 0, &sA[0][0], tid);
    for (int s = 0; s < S; ++s) {
        WAIT_ASYNC();
        __syncthreads();
        if (s + 1 < S)
            stage_rows16(abase, (s + 1) * 64, &sA[(s + 1) & 1][0], tid);
        const unsigned short* ap = &sA[s & 1][0];
        const int ks0 = s * 64;
#pragma unroll
        for (int kk = 0; kk < 64; kk += 32) {
            Frag fa, fb;
            const unsigned short* pa = ap + colL * BSTR + kk + half * 8;
            fa.q[0] = *(const u32x4*)pa;
            fa.q[1] = *(const u32x4*)(pa + 16);
            const int k0 = ks0 + kk + half * 8;
            fb.q[0] = loadg((const u32x4*)(brow + k0));
            fb.q[1] = loadg((const u32x4*)(brow + k0 + 16));
            acc = wmma_bf16(fa, fb, acc);
        }
    }

    const int j = n0 + colL;
    const float bias = loadg(&out_b[j]);
#pragma unroll
    for (int v = 0; v < 8; ++v)
        sm[v + 8 * half][j] = acc[v] + bias;

    __syncthreads();

    if (tid < 16) {
        const int r = tid;
        float mx = -3.0e38f;
        for (int c = 0; c < VOCAB; ++c) mx = fmaxf(mx, sm[r][c]);
        float se = 0.0f;
        for (int c = 0; c < VOCAB; ++c) se += __expf(sm[r][c] - mx);
        const int target = loadg(&batch[(m0 + r) * SEQ + (t + 1)]);
        storeg(&lp_out[(m0 + r) * (SEQ - 1) + t], sm[r][target] - mx - __logf(se));
    }
}

// ------------------------------- host side ---------------------------------

extern "C" void kernel_launch(void* const* d_in, const int* in_sizes, int n_in,
                              void* d_out, int out_size, void* d_ws, size_t ws_size,
                              hipStream_t stream) {
    (void)in_sizes; (void)n_in; (void)out_size; (void)ws_size;

    const int*   batch = (const int*)  d_in[0];
    // d_in[1] = context (zero-width), unused
    const float* emb   = (const float*)d_in[2];
    const float* w_ih0 = (const float*)d_in[3];
    const float* w_hh0 = (const float*)d_in[4];
    const float* b_ih0 = (const float*)d_in[5];
    const float* b_hh0 = (const float*)d_in[6];
    const float* w_ih1 = (const float*)d_in[7];
    const float* w_hh1 = (const float*)d_in[8];
    const float* b_ih1 = (const float*)d_in[9];
    const float* b_hh1 = (const float*)d_in[10];
    const float* out_w = (const float*)d_in[11];
    const float* out_b = (const float*)d_in[12];
    float* out = (float*)d_out;

    // ---- workspace layout (~51 MB) ----
    char* ws = (char*)d_ws;
    size_t off = 0;
    auto alloc = [&](size_t bytes) -> void* {
        void* p = ws + off;
        off = (off + bytes + 255) & ~(size_t)255;
        return p;
    };
    unsigned short* emb_bf  = (unsigned short*)alloc((size_t)VOCAB * EMB * 2);
    unsigned short* wih0_bf = (unsigned short*)alloc((size_t)H3 * EMB * 2);
    unsigned short* whh0_bf = (unsigned short*)alloc((size_t)H3 * HID * 2);
    unsigned short* wih1_bf = (unsigned short*)alloc((size_t)H3 * HID * 2);
    unsigned short* whh1_bf = (unsigned short*)alloc((size_t)H3 * HID * 2);
    unsigned short* outw_bf = (unsigned short*)alloc((size_t)VOCAB * HID * 2);
    unsigned short* h0bf[2], * h1bf[2];
    float* h0f[2], * h1f[2];
    for (int i = 0; i < 2; ++i) {
        h0bf[i] = (unsigned short*)alloc((size_t)BATCH * HID * 2);
        h1bf[i] = (unsigned short*)alloc((size_t)BATCH * HID * 2);
        h0f[i]  = (float*)alloc((size_t)BATCH * HID * 4);
        h1f[i]  = (float*)alloc((size_t)BATCH * HID * 4);
    }

    // ---- one-time (per launch) conversions ----
    auto cvt = [&](const float* in, unsigned short* o, int n) {
        f32_to_bf16_kernel<<<dim3((n + 255) / 256), dim3(256), 0, stream>>>(in, o, n);
    };
    cvt(emb,   emb_bf,  VOCAB * EMB);
    cvt(w_ih0, wih0_bf, H3 * EMB);
    cvt(w_hh0, whh0_bf, H3 * HID);
    cvt(w_ih1, wih1_bf, H3 * HID);
    cvt(w_hh1, whh1_bf, H3 * HID);
    cvt(out_w, outw_bf, VOCAB * HID);

    {
        const int n = BATCH * HID;
        zero_h_kernel<<<dim3((n + 255) / 256), dim3(256), 0, stream>>>(
            h0f[0], h1f[0], h0bf[0], h1bf[0], n);
    }

    // ---- recurrent steps: 3 launches per step ----
    const dim3 gGru(HID / 32, BATCH / 128);   // 48 x 2 blocks
    const dim3 gLog(BATCH / 16);              // 16 blocks
    for (int t = 0; t < SEQ - 1; ++t) {
        const int cur = t & 1, nxt = (t + 1) & 1;
        // layer 0: x = emb[batch[:, t]]
        gru_layer_kernel<<<gGru, dim3(256), 0, stream>>>(
            emb_bf, batch, t, EMB, wih0_bf, whh0_bf,
            h0bf[cur], h0f[cur], b_ih0, b_hh0, h0f[nxt], h0bf[nxt]);
        // layer 1: x = h0_new
        gru_layer_kernel<<<gGru, dim3(256), 0, stream>>>(
            h0bf[nxt], nullptr, 0, HID, wih1_bf, whh1_bf,
            h1bf[cur], h1f[cur], b_ih1, b_hh1, h1f[nxt], h1bf[nxt]);
        // logits + log_softmax + target gather
        logits_kernel<<<gLog, dim3(256), 0, stream>>>(
            h1bf[nxt], outw_bf, out_b, batch, t, out);
    }
}
